// SS2DBlock_27943057228199
// MI455X (gfx1250) — compile-verified, matrix operand order checked by reading, not run on hardware
//
#include <hip/hip_runtime.h>
#include <hip/hip_bf16.h>

typedef __attribute__((ext_vector_type(16))) __bf16 v16bf;
typedef __attribute__((ext_vector_type(8)))  float  v8f;

#define DIM  96
#define HID  192
#define NST  16
#define RANK 12
#define HW   4096          // 64*64
#define M1   16384         // B(4) * HW
#define MT   65536         // 4 dirs * M1

#if defined(__HIP_DEVICE_COMPILE__) && __has_builtin(__builtin_amdgcn_global_load_async_to_lds_b32)
#define HAVE_ASYNC_LDS 1
#endif

__device__ __forceinline__ float siluf(float v) {
    return v * (1.0f / (1.0f + __expf(-v)));
}

#ifdef HAVE_ASYNC_LDS
// async copy of one dword global -> LDS (tracked by ASYNCcnt)
__device__ __forceinline__ void async_g2l_b32(const float* g, float* l) {
    __builtin_amdgcn_global_load_async_to_lds_b32(
        (__attribute__((address_space(1))) int*)(uintptr_t)g,
        (__attribute__((address_space(3))) int*)(unsigned)(uintptr_t)l,
        0, 0);
}
#endif

// ---------------------------------------------------------------------------
// Generic bf16 WMMA GEMM:  C[M,N] = act(A[M,K] @ W[N,K]^T + bias)
// Each wave owns TWO adjacent 16x16 N-tiles: one shared A fragment feeds two
// independent v_wmma_f32_16x16x32_bf16 accumulator chains. The uniform
// "two tiles?" decision is hoisted out of the K-loop so the steady-state
// inner loop is branch-free (no per-iteration exec masking around WMMA).
// A rows optionally gathered via rowmapA; C rows optionally scattered via
// rowmapC with a column offset.
// ---------------------------------------------------------------------------
__global__ void wmma_gemm(const __hip_bfloat16* __restrict__ A,
                          const int* __restrict__ rowmapA,
                          int M, int K,
                          const __hip_bfloat16* __restrict__ W, int N,
                          const float* __restrict__ bias, int act,
                          float* __restrict__ Cf, __hip_bfloat16* __restrict__ Cb,
                          const int* __restrict__ rowmapC, int coloff, int ldc) {
    const int lane   = threadIdx.x & 31;
    const int wid    = threadIdx.x >> 5;
    const int Ntiles = N >> 4;
    const int nt0    = (blockIdx.x * 4 + wid) * 2;   // wave-uniform
    if (nt0 >= Ntiles) return;                       // whole-wave exit: EXEC stays full
    const bool two = (nt0 + 1) < Ntiles;             // wave-uniform

    const int row0  = blockIdx.y << 4;
    const int l15   = lane & 15;
    const int khalf = (lane >> 4) << 4;              // 0 or 16
    const int arow  = row0 + l15;
    const int asrc  = rowmapA ? rowmapA[arow] : arow;
    const int col0  = (nt0 << 4) + l15;

    const __hip_bfloat16* Ap  = A + (size_t)asrc * K + khalf;
    const __hip_bfloat16* Wp0 = W + (size_t)col0 * K + khalf;
    const __hip_bfloat16* Wp1 = Wp0 + (size_t)16 * K;

    v8f acc0 = {}, acc1 = {};
    if (two) {
        for (int k0 = 0; k0 < K; k0 += 32) {
            v16bf a  = *(const v16bf*)(Ap  + k0);
            v16bf b0 = *(const v16bf*)(Wp0 + k0);
            v16bf b1 = *(const v16bf*)(Wp1 + k0);
            if (k0 + 32 < K) {
                __builtin_prefetch(Ap  + k0 + 32, 0, 3);   // global_prefetch_b8
                __builtin_prefetch(Wp0 + k0 + 32, 0, 3);
                __builtin_prefetch(Wp1 + k0 + 32, 0, 3);
            }
            acc0 = __builtin_amdgcn_wmma_f32_16x16x32_bf16(
                       false, a, false, b0, (short)0, acc0, false, false);
            acc1 = __builtin_amdgcn_wmma_f32_16x16x32_bf16(
                       false, a, false, b1, (short)0, acc1, false, false);
        }
    } else {
        for (int k0 = 0; k0 < K; k0 += 32) {
            v16bf a  = *(const v16bf*)(Ap  + k0);
            v16bf b0 = *(const v16bf*)(Wp0 + k0);
            if (k0 + 32 < K) {
                __builtin_prefetch(Ap  + k0 + 32, 0, 3);
                __builtin_prefetch(Wp0 + k0 + 32, 0, 3);
            }
            acc0 = __builtin_amdgcn_wmma_f32_16x16x32_bf16(
                       false, a, false, b0, (short)0, acc0, false, false);
        }
    }

    const int rbase = row0 + ((lane >> 4) << 3);   // C/D layout: VGPR r -> M = r + 8*(lane/16)
    const float bv0 = bias ? bias[col0] : 0.0f;
#pragma unroll
    for (int r = 0; r < 8; ++r) {
        float v = acc0[r] + bv0;
        if (act == 1) v = siluf(v);
        const int m    = rbase + r;
        const int orow = rowmapC ? rowmapC[m] : m;
        const size_t o = (size_t)orow * ldc + coloff + col0;
        if (Cb) Cb[o] = __float2bfloat16(v);
        else    Cf[o] = v;
    }
    if (two) {
        const int col1  = col0 + 16;
        const float bv1 = bias ? bias[col1] : 0.0f;
#pragma unroll
        for (int r = 0; r < 8; ++r) {
            float v = acc1[r] + bv1;
            if (act == 1) v = siluf(v);
            const int m    = rbase + r;
            const int orow = rowmapC ? rowmapC[m] : m;
            const size_t o = (size_t)orow * ldc + coloff + col1;
            if (Cb) Cb[o] = __float2bfloat16(v);
            else    Cf[o] = v;
        }
    }
}

// ---------------------------------------------------------------------------
// Elementwise / prep kernels
// ---------------------------------------------------------------------------
__global__ void cvt_bf16(const float* __restrict__ s, __hip_bfloat16* __restrict__ d, int n) {
    int i = blockIdx.x * blockDim.x + threadIdx.x;
    if (i < n) d[i] = __float2bfloat16(s[i]);
}

// pad Wx (4,44,192) -> (4,48,192) bf16, zero rows 44..47
__global__ void wx_pad(const float* __restrict__ s, __hip_bfloat16* __restrict__ d) {
    int i = blockIdx.x * blockDim.x + threadIdx.x;
    if (i >= 4 * 48 * HID) return;
    int k = i % HID, n = (i / HID) % 48, dd = i / (HID * 48);
    d[i] = (n < 44) ? __float2bfloat16(s[((size_t)dd * 44 + n) * HID + k])
                    : __float2bfloat16(0.0f);
}

// rowmap[m] for m = (dir,b,l): pixel row (b*4096 + p) in image order
__global__ void build_rowmap(int* __restrict__ rm) {
    int m = blockIdx.x * blockDim.x + threadIdx.x;
    if (m >= MT) return;
    int d = m >> 14, i = m & 16383, b = i >> 12, l = i & 4095;
    int p;
    if      (d == 0) p = l;
    else if (d == 1) p = 4095 - l;
    else if (d == 2) p = ((l & 63) << 6) | (l >> 6);
    else { int l2 = 4095 - l; p = ((l2 & 63) << 6) | (l2 >> 6); }
    rm[m] = (b << 12) | p;
}

// LayerNorm over C=96 of NCHW input -> bf16 rows [M1,96]
__global__ void ln1_kernel(const float* __restrict__ x, const float* __restrict__ g,
                           const float* __restrict__ b, __hip_bfloat16* __restrict__ xn) {
    int r = blockIdx.x * blockDim.x + threadIdx.x;
    if (r >= M1) return;
    int bb = r >> 12, p = r & 4095;
    const float* xp = x + (size_t)bb * DIM * HW + p;
    float s = 0.f, s2 = 0.f;
    for (int c = 0; c < DIM; ++c) { float v = xp[(size_t)c * HW]; s += v; s2 += v * v; }
    float mu  = s * (1.0f / DIM);
    float var = s2 * (1.0f / DIM) - mu * mu;
    float rs  = rsqrtf(var + 1e-5f);
    __hip_bfloat16* o = xn + (size_t)r * DIM;
    for (int c = 0; c < DIM; ++c)
        o[c] = __float2bfloat16((xp[(size_t)c * HW] - mu) * rs * g[c] + b[c]);
}

// depthwise 3x3 conv (SAME) + bias + silu on (b,h,w,c) f32 -> bf16
__global__ void dwconv_kernel(const float* __restrict__ in, const float* __restrict__ w,
                              const float* __restrict__ bias, __hip_bfloat16* __restrict__ out) {
    int i = blockIdx.x * blockDim.x + threadIdx.x;
    if (i >= M1 * HID) return;
    int c = i % HID, r = i / HID, bb = r >> 12, p = r & 4095, h = p >> 6, ww = p & 63;
    float acc = bias[c];
    const float* wc = w + c * 9;
#pragma unroll
    for (int ky = 0; ky < 3; ++ky) {
        int hh = h + ky - 1;
        if ((unsigned)hh >= 64u) continue;
#pragma unroll
        for (int kx = 0; kx < 3; ++kx) {
            int xx = ww + kx - 1;
            if ((unsigned)xx >= 64u) continue;
            acc += in[((size_t)(bb << 12) + hh * 64 + xx) * HID + c] * wc[ky * 3 + kx];
        }
    }
    out[i] = __float2bfloat16(siluf(acc));
}

// causal conv1d K=4 over seq + silu; per-direction weights; writes f32 + bf16
__global__ void conv1d_kernel(const __hip_bfloat16* __restrict__ x,
                              const float* __restrict__ cw, const float* __restrict__ cb,
                              float* __restrict__ xf, __hip_bfloat16* __restrict__ xb) {
    int i = blockIdx.x * blockDim.x + threadIdx.x;
    if (i >= MT * HID) return;
    int c = i % HID, m = i / HID, d = m >> 14, t = m & 4095;
    const float* wk = cw + ((size_t)d * HID + c) * 4;
    float acc = cb[d * HID + c];
#pragma unroll
    for (int k = 0; k < 4; ++k) {
        int tt = t + k - 3;
        if (tt >= 0) acc += __bfloat162float(x[(size_t)(m + k - 3) * HID + c]) * wk[k];
    }
    float v = siluf(acc);
    xf[i] = v;
    xb[i] = __float2bfloat16(v);
}

// Selective scan: one block per (dir,batch) sequence, one thread per channel.
// Fuses dt = softplus(dbl[:, :12] @ Wdt^T + bdt). State h[16] in VGPRs.
// The shared 48-float row (dt-in | B | C) is double-buffered in LDS and
// fetched with GLOBAL_LOAD_ASYNC_TO_LDS_B32 one step ahead (ASYNCcnt),
// overlapping memory latency with the recurrence on the serial critical path.
__global__ void scan_kernel(const float* __restrict__ dbl, const float* __restrict__ xc,
                            const float* __restrict__ Wdt, const float* __restrict__ bdt,
                            const float* __restrict__ Alog, const float* __restrict__ Dp,
                            float* __restrict__ y) {
    const int blk = blockIdx.x;        // d*4 + b
    const int d   = blk >> 2;
    const int c   = threadIdx.x;       // 0..191
    const int dc  = d * HID + c;

    float A2[NST];                     // A * log2(e): use native base-2 v_exp_f32
#pragma unroll
    for (int n = 0; n < NST; ++n)
        A2[n] = -__expf(Alog[(size_t)dc * NST + n]) * 1.4426950408889634f;
    float wdt[RANK];
#pragma unroll
    for (int r = 0; r < RANK; ++r) wdt[r] = Wdt[(size_t)dc * RANK + r];
    const float bd = bdt[dc];
    const float Dc = Dp[dc];

    float h[NST];
#pragma unroll
    for (int n = 0; n < NST; ++n) h[n] = 0.0f;

    __shared__ float srow[2][48];
    const size_t mbase = (size_t)blk * 4096;

#ifdef HAVE_ASYNC_LDS
    if (c < 48) async_g2l_b32(&dbl[mbase * 48 + c], &srow[0][c]);
    asm volatile("s_wait_asynccnt 0x0" ::: "memory");
    __syncthreads();
#endif

    for (int t = 0; t < 4096; ++t) {
        const size_t m = mbase + t;
#ifdef HAVE_ASYNC_LDS
        const float* cur = srow[t & 1];
        if ((t + 1 < 4096) && (c < 48))
            async_g2l_b32(&dbl[(m + 1) * 48 + c], &srow[(t + 1) & 1][c]);
#else
        __syncthreads();
        if (c < 48) srow[0][c] = dbl[m * 48 + c];
        __syncthreads();
        const float* cur = srow[0];
#endif
        float a = bd;
#pragma unroll
        for (int r = 0; r < RANK; ++r) a += cur[r] * wdt[r];
        const float dt = (a > 20.0f) ? a : log1pf(__expf(a));
        const float u  = xc[m * HID + c];
        const float du = dt * u;
        float yv = 0.0f;
#pragma unroll
        for (int n = 0; n < NST; ++n) {
            h[n] = exp2f(dt * A2[n]) * h[n] + du * cur[12 + n];
            yv  += h[n] * cur[28 + n];
        }
        y[m * HID + c] = yv + u * Dc;

#ifdef HAVE_ASYNC_LDS
        asm volatile("s_wait_asynccnt 0x0" ::: "memory");
        __syncthreads();
#endif
    }
}

// y * silu(z)  (zs already holds silu(z) in bf16) -> bf16
__global__ void gatez_kernel(const float* __restrict__ y, const __hip_bfloat16* __restrict__ zs,
                             __hip_bfloat16* __restrict__ o) {
    int i = blockIdx.x * blockDim.x + threadIdx.x;
    if (i < MT * HID) o[i] = __float2bfloat16(y[i] * __bfloat162float(zs[i]));
}

// LayerNorm over 192 + multiply by precomputed gate -> bf16
__global__ void ln2gate_kernel(const float* __restrict__ ss, const float* __restrict__ g,
                               const float* __restrict__ b, const float* __restrict__ gate,
                               __hip_bfloat16* __restrict__ o) {
    int r = blockIdx.x * blockDim.x + threadIdx.x;
    if (r >= M1) return;
    const float* sp = ss + (size_t)r * HID;
    float s = 0.f, s2 = 0.f;
    for (int c = 0; c < HID; ++c) { float v = sp[c]; s += v; s2 += v * v; }
    float mu  = s * (1.0f / HID);
    float var = s2 * (1.0f / HID) - mu * mu;
    float rs  = rsqrtf(var + 1e-5f);
    __hip_bfloat16* op = o + (size_t)r * HID;
    const float* gp = gate + (size_t)r * HID;
    for (int c = 0; c < HID; ++c)
        op[c] = __float2bfloat16(((sp[c] - mu) * rs * g[c] + b[c]) * gp[c]);
}

// add residual, write NCHW f32
__global__ void final_kernel(const float* __restrict__ o, const float* __restrict__ x,
                             float* __restrict__ out) {
    int i = blockIdx.x * blockDim.x + threadIdx.x;
    if (i >= 4 * DIM * HW) return;
    int p = i & 4095, c = (i >> 12) % DIM, b = i / (DIM * HW);
    out[i] = o[((size_t)(b << 12) + p) * DIM + c] + x[i];
}

// ---------------------------------------------------------------------------
static inline void gemm(hipStream_t s, const __hip_bfloat16* A, const int* rmA,
                        int M, int K, const __hip_bfloat16* W, int N,
                        const float* bias, int act, float* Cf, __hip_bfloat16* Cb,
                        const int* rmC, int coloff, int ldc) {
    dim3 grid(((N >> 4) + 7) / 8, M >> 4);   // 4 waves/block, 2 N-tiles/wave
    wmma_gemm<<<grid, 128, 0, s>>>(A, rmA, M, K, W, N, bias, act, Cf, Cb, rmC, coloff, ldc);
}

extern "C" void kernel_launch(void* const* d_in, const int* in_sizes, int n_in,
                              void* d_out, int out_size, void* d_ws, size_t ws_size,
                              hipStream_t stream) {
    const float* x      = (const float*)d_in[0];
    const float* ln1_g  = (const float*)d_in[1];
    const float* ln1_b  = (const float*)d_in[2];
    const float* Wm     = (const float*)d_in[3];
    const float* bm     = (const float*)d_in[4];
    const float* Wg     = (const float*)d_in[5];
    const float* bg     = (const float*)d_in[6];
    const float* dw_w   = (const float*)d_in[7];
    const float* dw_b   = (const float*)d_in[8];
    const float* m_Win  = (const float*)d_in[9];
    const float* m_cw   = (const float*)d_in[10];
    const float* m_cb   = (const float*)d_in[11];
    const float* m_Wx   = (const float*)d_in[12];
    const float* m_Wdt  = (const float*)d_in[13];
    const float* m_bdt  = (const float*)d_in[14];
    const float* m_Alog = (const float*)d_in[15];
    const float* m_D    = (const float*)d_in[16];
    const float* m_Wout = (const float*)d_in[17];
    const float* so_W   = (const float*)d_in[18];
    const float* so_b   = (const float*)d_in[19];
    const float* ln2_g  = (const float*)d_in[20];
    const float* ln2_b  = (const float*)d_in[21];
    const float* out_W  = (const float*)d_in[22];
    const float* out_b  = (const float*)d_in[23];
    float* out = (float*)d_out;

    // ---- workspace arena (bump allocator, 256B aligned) ----
    char* base = (char*)d_ws;
    size_t off = 0;
    auto alloc = [&](size_t bytes) -> char* {
        char* p = base + off;
        off += (bytes + 255) & ~(size_t)255;
        return p;
    };
    typedef __hip_bfloat16 bf;
    bf*  wWm   = (bf*)alloc(HID * DIM * 2);
    bf*  wWg   = (bf*)alloc(HID * DIM * 2);
    bf*  wWin  = (bf*)alloc((size_t)4 * 384 * HID * 2);
    bf*  wWx   = (bf*)alloc((size_t)4 * 48 * HID * 2);
    bf*  wWout = (bf*)alloc((size_t)4 * HID * HID * 2);
    bf*  wSo   = (bf*)alloc((size_t)HID * 768 * 2);
    bf*  wOutW = (bf*)alloc((size_t)DIM * HID * 2);
    int* rmap  = (int*)alloc((size_t)MT * 4);
    bf*  xn    = (bf*)alloc((size_t)M1 * DIM * 2);
    float* gateb = (float*)alloc((size_t)M1 * HID * 4);
    float* mainp = (float*)alloc((size_t)M1 * HID * 4);
    bf*  maincb  = (bf*)alloc((size_t)M1 * HID * 2);
    bf*  xbf     = (bf*)alloc((size_t)MT * HID * 2);
    bf*  zsbf    = (bf*)alloc((size_t)MT * HID * 2);
    float* xcf   = (float*)alloc((size_t)MT * HID * 4);
    bf*  xcbf    = (bf*)alloc((size_t)MT * HID * 2);   // later reused for y*silu(z)
    float* dblf  = (float*)alloc((size_t)MT * 48 * 4);
    float* yf    = (float*)alloc((size_t)MT * HID * 4);
    bf*  ycat    = (bf*)alloc((size_t)M1 * 768 * 2);
    float* ssf   = (float*)alloc((size_t)M1 * HID * 4);
    bf*  ssgb    = (bf*)alloc((size_t)M1 * HID * 2);
    float* of    = (float*)alloc((size_t)M1 * DIM * 4);
    (void)ws_size; (void)in_sizes; (void)n_in; (void)out_size;

    const int TB = 256;
    auto grid1 = [&](size_t n) { return dim3((unsigned)((n + TB - 1) / TB)); };

    // ---- weight staging (f32 -> bf16) ----
    cvt_bf16<<<grid1(HID * DIM), TB, 0, stream>>>(Wm, wWm, HID * DIM);
    cvt_bf16<<<grid1(HID * DIM), TB, 0, stream>>>(Wg, wWg, HID * DIM);
    cvt_bf16<<<grid1(4 * 384 * HID), TB, 0, stream>>>(m_Win, wWin, 4 * 384 * HID);
    wx_pad  <<<grid1(4 * 48 * HID), TB, 0, stream>>>(m_Wx, wWx);
    cvt_bf16<<<grid1(4 * HID * HID), TB, 0, stream>>>(m_Wout, wWout, 4 * HID * HID);
    cvt_bf16<<<grid1(HID * 768), TB, 0, stream>>>(so_W, wSo, HID * 768);
    cvt_bf16<<<grid1(DIM * HID), TB, 0, stream>>>(out_W, wOutW, DIM * HID);
    build_rowmap<<<grid1(MT), TB, 0, stream>>>(rmap);

    // ---- LN1 + input projections ----
    ln1_kernel<<<grid1(M1), TB, 0, stream>>>(x, ln1_g, ln1_b, xn);
    gemm(stream, xn, nullptr, M1, DIM, wWm, HID, bm, 0, mainp, nullptr, nullptr, 0, HID);
    gemm(stream, xn, nullptr, M1, DIM, wWg, HID, bg, 1, gateb, nullptr, nullptr, 0, HID);

    // ---- depthwise 3x3 + silu ----
    dwconv_kernel<<<grid1((size_t)M1 * HID), TB, 0, stream>>>(mainp, dw_w, dw_b, maincb);

    // ---- per-direction in-projection (gathered A rows; x part + silu(z) part) ----
    for (int d = 0; d < 4; ++d) {
        const bf* Wd = wWin + (size_t)d * 384 * HID;
        gemm(stream, maincb, rmap + (size_t)d * M1, M1, HID, Wd, HID,
             nullptr, 0, nullptr, xbf + (size_t)d * M1 * HID, nullptr, 0, HID);
        gemm(stream, maincb, rmap + (size_t)d * M1, M1, HID, Wd + (size_t)HID * HID, HID,
             nullptr, 1, nullptr, zsbf + (size_t)d * M1 * HID, nullptr, 0, HID);
    }

    // ---- causal conv1d + silu ----
    conv1d_kernel<<<grid1((size_t)MT * HID), TB, 0, stream>>>(xbf, m_cw, m_cb, xcf, xcbf);

    // ---- x-projection (dt-input | B | C), padded N=48 ----
    for (int d = 0; d < 4; ++d)
        gemm(stream, xcbf + (size_t)d * M1 * HID, nullptr, M1, HID,
             wWx + (size_t)d * 48 * HID, 48, nullptr, 0,
             dblf + (size_t)d * M1 * 48, nullptr, nullptr, 0, 48);

    // ---- selective scan (sequential over L=4096, async double-buffered) ----
    scan_kernel<<<16, HID, 0, stream>>>(dblf, xcf, m_Wdt, m_bdt, m_Alog, m_D, yf);

    // ---- gate by silu(z), out-projection scattered into ycat ----
    gatez_kernel<<<grid1((size_t)MT * HID), TB, 0, stream>>>(yf, zsbf, xcbf);
    for (int d = 0; d < 4; ++d)
        gemm(stream, xcbf + (size_t)d * M1 * HID, nullptr, M1, HID,
             wWout + (size_t)d * HID * HID, HID, nullptr, 0,
             nullptr, ycat, rmap + (size_t)d * M1, d * HID, 768);

    // ---- output projection chain ----
    gemm(stream, ycat, nullptr, M1, 768, wSo, HID, so_b, 0, ssf, nullptr, nullptr, 0, HID);
    ln2gate_kernel<<<grid1(M1), TB, 0, stream>>>(ssf, ln2_g, ln2_b, gateb, ssgb);
    gemm(stream, ssgb, nullptr, M1, HID, wOutW, DIM, out_b, 0, of, nullptr, nullptr, 0, DIM);
    final_kernel<<<grid1((size_t)4 * DIM * HW), TB, 0, stream>>>(of, x, out);
}
